// MambaHead_65189013618748
// MI455X (gfx1250) — compile-verified, matrix operand order checked by reading, not run on hardware
//
#include <hip/hip_runtime.h>
#include <hip/hip_bf16.h>

// ---------------------------------------------------------------------------
// Mamba block forward for MI455X (gfx1250, wave32, WMMA + async LDS copies).
// B=4, L=2048, DM=1024, DI=2048, DS=16, DC=4, DTR=64.  BL = B*L = 8192.
// GEMMs: bf16 WMMA (f32 accum), global->LDS staging via
// GLOBAL_LOAD_ASYNC_TO_LDS_B128 with double-buffered LDS tiles.
// ---------------------------------------------------------------------------

#define BB  4
#define LL  2048
#define DM  1024
#define DI  2048
#define DSZ 16
#define DTR 64
#define BL  (BB * LL)

typedef __attribute__((ext_vector_type(16))) __bf16 v16bf;
typedef __attribute__((ext_vector_type(8)))  __bf16 v8bf;
typedef __attribute__((ext_vector_type(8)))  float  v8f;

union Frag { v16bf v; v8bf h[2]; };

static __device__ __forceinline__ unsigned short f32_to_bf16(float f) {
  unsigned int u = __float_as_uint(f);
  u += 0x7fffu + ((u >> 16) & 1u);  // round-to-nearest-even
  return (unsigned short)(u >> 16);
}

static __device__ __forceinline__ float silu(float x) {
  return x / (1.f + __expf(-x));
}

// Async 16-byte global -> LDS copy (CDNA5 TDM-adjacent path, ASYNCcnt).
// VDST vgpr holds the LDS byte address, vaddr pair holds the global address.
static __device__ __forceinline__ void async_copy16(const void* gptr, void* lptr) {
  unsigned int l = (unsigned int)(unsigned long long)(uintptr_t)lptr;
  unsigned long long g = (unsigned long long)(uintptr_t)gptr;
  asm volatile("global_load_async_to_lds_b128 %0, %1, off"
               :: "v"(l), "v"(g) : "memory");
}

static __device__ __forceinline__ void wait_async0() {
  asm volatile("s_wait_asynccnt 0x0" ::: "memory");
}

// ---------------------------------------------------------------------------
// Generic tiled GEMM:  C[M,N] = A[M,K] * W[N,K]^T, bf16 in (ushort storage),
// f32 out.  Requires M % 128 == 0, K % 32 == 0 (true for all call sites).
// Block tile 128(M) x 256(N), 8 waves, each wave a 64x64 tile (4x4 WMMA).
// Double-buffered LDS, async global->LDS copies overlapped with WMMA.
// ---------------------------------------------------------------------------
#define LDA 40
#define LDB 40

__global__ __launch_bounds__(256)
void gemm_bf16_wmma(const unsigned short* __restrict__ A,
                    const unsigned short* __restrict__ W,
                    float* __restrict__ C,
                    int M, int N, int K) {
  __shared__ __align__(16) unsigned short As[2][128 * LDA];
  __shared__ __align__(16) unsigned short Bs[2][256 * LDB];

  const int tid  = threadIdx.x;
  const int bm   = blockIdx.y;
  const int bn   = blockIdx.x;
  const int wave = tid >> 5;
  const int wm   = wave & 1;   // 0..1  -> M offset
  const int wn   = wave >> 1;  // 0..3  -> N offset
  const int lane = tid & 31;
  const int lr   = lane & 15;
  const int c0   = (lane >> 4) << 3;  // 0 or 8 : per-lane K-chunk base
  const int nk   = K >> 5;

  v8f acc[4][4] = {};

  // Issue async copies for K-tile kt into LDS buffer `buf`.
  auto issue = [&](int kt, int buf) {
    const int k0 = kt << 5;
    {  // A tile: 128 rows x 32 K; each thread 2x16B
      int r  = tid >> 1;
      int kc = (tid & 1) << 4;
      const unsigned short* src = A + (size_t)(bm * 128 + r) * K + k0 + kc;
      async_copy16(src,     &As[buf][r * LDA + kc]);
      async_copy16(src + 8, &As[buf][r * LDA + kc + 8]);
    }
    {  // B tile: 256 N-rows x 32 K; each thread 4x16B (guarded)
      int gn = bn * 256 + tid;
      if (gn < N) {
        const unsigned short* src = W + (size_t)gn * K + k0;
        #pragma unroll
        for (int c = 0; c < 32; c += 8)
          async_copy16(src + c, &Bs[buf][tid * LDB + c]);
      } else {
        uint4 z = {0u, 0u, 0u, 0u};
        #pragma unroll
        for (int c = 0; c < 32; c += 8)
          *(uint4*)&Bs[buf][tid * LDB + c] = z;
      }
    }
  };

  issue(0, 0);

  for (int kt = 0; kt < nk; ++kt) {
    const int buf = kt & 1;
    wait_async0();       // this wave's tile-kt copies have landed in LDS
    __syncthreads();     // everyone's have; previous reads of buf^1 retired
    if (kt + 1 < nk) issue(kt + 1, buf ^ 1);  // overlap fetch with compute

    Frag af[4], bfr[4];
    #pragma unroll
    for (int i = 0; i < 4; ++i) {
      int row = wm * 64 + i * 16 + lr;
      af[i].h[0] = *(const v8bf*)&As[buf][row * LDA + c0];
      af[i].h[1] = *(const v8bf*)&As[buf][row * LDA + c0 + 16];
    }
    #pragma unroll
    for (int j = 0; j < 4; ++j) {
      int col = wn * 64 + j * 16 + lr;
      bfr[j].h[0] = *(const v8bf*)&Bs[buf][col * LDB + c0];
      bfr[j].h[1] = *(const v8bf*)&Bs[buf][col * LDB + c0 + 16];
    }
    #pragma unroll
    for (int i = 0; i < 4; ++i)
      #pragma unroll
      for (int j = 0; j < 4; ++j)
        acc[i][j] = __builtin_amdgcn_wmma_f32_16x16x32_bf16(
            false, af[i].v, false, bfr[j].v, (short)0, acc[i][j], false, false);
  }

  // Store: f32 C/D layout (elem e -> row e + 8*(lane>>4), col lane&15).
  const int half = lane >> 4;
  #pragma unroll
  for (int i = 0; i < 4; ++i)
    #pragma unroll
    for (int j = 0; j < 4; ++j) {
      int col = bn * 256 + wn * 64 + j * 16 + lr;
      if (col < N) {
        #pragma unroll
        for (int e = 0; e < 8; ++e) {
          int row = bm * 128 + wm * 64 + i * 16 + e + 8 * half;
          C[(size_t)row * N + col] = acc[i][j][e];
        }
      }
    }
}

// ---------------------------------------------------------------------------
// RMSNorm over DM=1024 columns, output bf16 (for GEMM A operand).
// ---------------------------------------------------------------------------
__global__ __launch_bounds__(256)
void rmsnorm_to_bf16(const float* __restrict__ x, const float* __restrict__ w,
                     unsigned short* __restrict__ out, int cols) {
  __shared__ float red[256];
  const int row = blockIdx.x, tid = threadIdx.x;
  const float* xr = x + (size_t)row * cols;
  float ss = 0.f;
  for (int c = tid; c < cols; c += 256) { float v = xr[c]; ss += v * v; }
  red[tid] = ss; __syncthreads();
  for (int s = 128; s > 0; s >>= 1) {
    if (tid < s) red[tid] += red[tid + s];
    __syncthreads();
  }
  const float rs = rsqrtf(red[0] / (float)cols + 1e-5f);
  for (int c = tid; c < cols; c += 256)
    out[(size_t)row * cols + c] = f32_to_bf16(xr[c] * rs * w[c]);
}

// f32 -> bf16(ushort) bulk convert (weights, activations)
__global__ void f2bf_kernel(const float* __restrict__ in,
                            unsigned short* __restrict__ out, long n) {
  long i = (long)blockIdx.x * blockDim.x + threadIdx.x;
  if (i < n) out[i] = f32_to_bf16(in[i]);
}

// ---------------------------------------------------------------------------
// Depthwise causal conv (DC=4) + bias + SiLU.  proj holds [BL, 2*DI]
// (u in cols 0..DI-1).  Writes f32 u and bf16 u (for x_proj GEMM).
// ---------------------------------------------------------------------------
__global__ void conv_silu_kernel(const float* __restrict__ proj,
                                 const float* __restrict__ cw,
                                 const float* __restrict__ cb,
                                 float* __restrict__ uc,
                                 unsigned short* __restrict__ ucb) {
  long idx = (long)blockIdx.x * blockDim.x + threadIdx.x;
  if (idx >= (long)BL * DI) return;
  int d     = (int)(idx & (DI - 1));
  int btrow = (int)(idx >> 11);       // DI == 2048
  int t     = btrow & (LL - 1);       // LL == 2048
  int bbase = btrow - t;              // b*L
  float acc = cb[d];
  #pragma unroll
  for (int k = 0; k < 4; ++k) {
    int tt = t + k - 3;
    if (tt >= 0)
      acc += proj[((size_t)(bbase + tt) << 12) + d] * cw[d * 4 + k]; // stride 4096
  }
  float s = silu(acc);
  uc[idx]  = s;
  ucb[idx] = f32_to_bf16(s);
}

// Extract dt rows (cols 0..63 of ssm[BL,96]) as bf16 GEMM input.
__global__ void extract_dt_kernel(const float* __restrict__ ssm,
                                  unsigned short* __restrict__ dti) {
  long idx = (long)blockIdx.x * blockDim.x + threadIdx.x;
  if (idx >= (long)BL * DTR) return;
  long row = idx >> 6;
  int  c   = (int)(idx & 63);
  dti[idx] = f32_to_bf16(ssm[row * 96 + c]);
}

// softplus(dtm + dt_proj_b), in place.
__global__ void softplus_bias_kernel(float* __restrict__ dtm,
                                     const float* __restrict__ bias) {
  long idx = (long)blockIdx.x * blockDim.x + threadIdx.x;
  if (idx >= (long)BL * DI) return;
  float v = dtm[idx] + bias[idx & (DI - 1)];
  dtm[idx] = (v > 20.f) ? v : log1pf(__expf(v));
}

// ---------------------------------------------------------------------------
// Selective scan.  One thread per (batch, channel d); 16-state in registers.
// B_t/C_t (shared across d) staged in LDS via async copies, 64-step chunks.
// u_y: in = conv'd u (f32), out = gated y, in place (read-before-write).
// ---------------------------------------------------------------------------
__global__ __launch_bounds__(256)
void mamba_scan_kernel(const float* __restrict__ dt,    // [BL, DI]
                       float* __restrict__ u_y,         // [BL, DI] in/out
                       const float* __restrict__ ssm,   // [BL, 96]
                       const float* __restrict__ proj,  // [BL, 2*DI] gate @ DI+d
                       const float* __restrict__ A_log, // [DI, 16]
                       const float* __restrict__ Dv) {  // [DI]
  const int tid = threadIdx.x;
  const int d   = blockIdx.x * 256 + tid;
  const int b   = blockIdx.y;
  __shared__ __align__(16) float sBC[64 * 32];  // per-t: [0..15]=B, [16..31]=C

  float Arow[16];
  #pragma unroll
  for (int s = 0; s < 16; ++s) Arow[s] = -__expf(A_log[d * 16 + s]);
  const float Dd = Dv[d];

  float st[16];
  #pragma unroll
  for (int s = 0; s < 16; ++s) st[s] = 0.f;

  for (int ch = 0; ch < LL / 64; ++ch) {
    // async-stage B/C for this chunk: 64 rows x 32 floats, 16B granules
    for (int i = tid; i < 64 * 8; i += 256) {
      int tt = i >> 3, q = i & 7;
      const float* src = ssm + (size_t)(b * LL + ch * 64 + tt) * 96 + DTR + q * 4;
      async_copy16(src, &sBC[tt * 32 + q * 4]);
    }
    // prefetch next chunk's dt/u rows while waiting
    if (ch + 1 < LL / 64) {
      __builtin_prefetch(&dt[(size_t)(b * LL + (ch + 1) * 64) * DI + d], 0, 0);
      __builtin_prefetch(&u_y[(size_t)(b * LL + (ch + 1) * 64) * DI + d], 0, 0);
    }
    wait_async0();
    __syncthreads();

    for (int tt = 0; tt < 64; ++tt) {
      size_t row = (size_t)(b * LL + ch * 64 + tt);
      float dtv = dt[row * DI + d];
      float uv  = u_y[row * DI + d];
      float du  = dtv * uv;
      float y   = 0.f;
      #pragma unroll
      for (int s = 0; s < 16; ++s) {
        float dA = __expf(dtv * Arow[s]);
        st[s] = dA * st[s] + du * sBC[tt * 32 + s];
        y += st[s] * sBC[tt * 32 + 16 + s];
      }
      float g = proj[(row << 12) + DI + d];
      u_y[row * DI + d] = (y + uv * Dd) * silu(g);
    }
    __syncthreads();  // chunk consumed before next async issue overwrites
  }
}

// ---------------------------------------------------------------------------
// Final: residual + rmsnorm on last token only, then 1024->8 head.
// One block per batch.
// ---------------------------------------------------------------------------
__global__ __launch_bounds__(256)
void final_head_kernel(const float* __restrict__ x,
                       const float* __restrict__ outm,
                       const float* __restrict__ nw,
                       const float* __restrict__ fcw,
                       const float* __restrict__ fcb,
                       float* __restrict__ logits) {
  __shared__ float sv[DM];
  __shared__ float red[256];
  __shared__ float po[8 * 256];
  const int b = blockIdx.x, tid = threadIdx.x;
  const size_t row = (size_t)b * LL + (LL - 1);
  float ss = 0.f;
  for (int c = tid; c < DM; c += 256) {
    float v = x[row * DM + c] + outm[row * DM + c];
    sv[c] = v; ss += v * v;
  }
  red[tid] = ss; __syncthreads();
  for (int s = 128; s > 0; s >>= 1) {
    if (tid < s) red[tid] += red[tid + s];
    __syncthreads();
  }
  const float rs = rsqrtf(red[0] / (float)DM + 1e-5f);
  float p[8];
  #pragma unroll
  for (int o = 0; o < 8; ++o) p[o] = 0.f;
  for (int c = tid; c < DM; c += 256) {
    float hv = sv[c] * rs * nw[c];
    #pragma unroll
    for (int o = 0; o < 8; ++o) p[o] += hv * fcw[o * DM + c];
  }
  #pragma unroll
  for (int o = 0; o < 8; ++o) po[o * 256 + tid] = p[o];
  __syncthreads();
  for (int s = 128; s > 0; s >>= 1) {
    if (tid < s) {
      #pragma unroll
      for (int o = 0; o < 8; ++o) po[o * 256 + tid] += po[o * 256 + tid + s];
    }
    __syncthreads();
  }
  if (tid < 8) logits[b * 8 + tid] = po[tid * 256] + fcb[tid];
}

// ---------------------------------------------------------------------------
// Host launcher.
// ---------------------------------------------------------------------------
extern "C" void kernel_launch(void* const* d_in, const int* in_sizes, int n_in,
                              void* d_out, int out_size, void* d_ws, size_t ws_size,
                              hipStream_t stream) {
  const float* x       = (const float*)d_in[0];   // [B,L,DM]
  const float* norm_w  = (const float*)d_in[1];   // [DM]
  const float* in_w    = (const float*)d_in[2];   // [2*DI, DM]
  const float* conv_w  = (const float*)d_in[3];   // [DI, 4]
  const float* conv_b  = (const float*)d_in[4];   // [DI]
  const float* xproj_w = (const float*)d_in[5];   // [96, DI]
  const float* dtp_w   = (const float*)d_in[6];   // [DI, DTR]
  const float* dtp_b   = (const float*)d_in[7];   // [DI]
  const float* A_log   = (const float*)d_in[8];   // [DI, DS]
  const float* Dv      = (const float*)d_in[9];   // [DI]
  const float* out_w   = (const float*)d_in[10];  // [DM, DI]
  const float* normf_w = (const float*)d_in[11];  // [DM]
  const float* fc_w    = (const float*)d_in[12];  // [8, DM]
  const float* fc_b    = (const float*)d_in[13];  // [8]
  float* logits = (float*)d_out;

  // Workspace bump allocator (256B aligned).
  char* base = (char*)d_ws;
  size_t off = 0;
  auto alloc = [&](size_t bytes) -> void* {
    void* p = base + off;
    off = (off + bytes + 255) & ~(size_t)255;
    return p;
  };
  unsigned short* hb   = (unsigned short*)alloc((size_t)BL * DM * 2);      // rmsnorm(x) bf16
  unsigned short* wi   = (unsigned short*)alloc((size_t)2 * DI * DM * 2);  // in_proj_w bf16
  float*          proj = (float*)alloc((size_t)BL * 2 * DI * 4);           // [BL, 4096]
  float*          uc   = (float*)alloc((size_t)BL * DI * 4);               // conv u (-> y in place)
  unsigned short* ucb  = (unsigned short*)alloc((size_t)BL * DI * 2);      // u bf16
  unsigned short* wx   = (unsigned short*)alloc((size_t)96 * DI * 2);      // x_proj_w bf16
  float*          ssm  = (float*)alloc((size_t)BL * 96 * 4);               // [BL, 96]
  unsigned short* dti  = (unsigned short*)alloc((size_t)BL * DTR * 2);     // dt input bf16
  unsigned short* wdt  = (unsigned short*)alloc((size_t)DI * DTR * 2);     // dt_proj_w bf16
  float*          dtm  = (float*)alloc((size_t)BL * DI * 4);               // dt (softplus in place)
  unsigned short* yb   = (unsigned short*)alloc((size_t)BL * DI * 2);      // y bf16
  unsigned short* wo   = (unsigned short*)alloc((size_t)DM * DI * 2);      // out_proj_w bf16
  float*          outm = (float*)alloc((size_t)BL * DM * 4);               // [BL, DM]
  (void)ws_size; (void)n_in; (void)in_sizes; (void)out_size;

  auto blocks1d = [](long n) { return (unsigned)((n + 255) / 256); };

  // 1. rmsnorm -> bf16
  rmsnorm_to_bf16<<<BL, 256, 0, stream>>>(x, norm_w, hb, DM);
  // 2. weight conversions
  f2bf_kernel<<<blocks1d((long)2 * DI * DM), 256, 0, stream>>>(in_w, wi, (long)2 * DI * DM);
  f2bf_kernel<<<blocks1d((long)96 * DI), 256, 0, stream>>>(xproj_w, wx, (long)96 * DI);
  f2bf_kernel<<<blocks1d((long)DI * DTR), 256, 0, stream>>>(dtp_w, wdt, (long)DI * DTR);
  f2bf_kernel<<<blocks1d((long)DM * DI), 256, 0, stream>>>(out_w, wo, (long)DM * DI);
  // 3. in_proj: [BL,DM] x [4096,DM]^T -> [BL,4096]
  gemm_bf16_wmma<<<dim3(2 * DI / 256, BL / 128), 256, 0, stream>>>(hb, wi, proj, BL, 2 * DI, DM);
  // 4. conv + silu
  conv_silu_kernel<<<blocks1d((long)BL * DI), 256, 0, stream>>>(proj, conv_w, conv_b, uc, ucb);
  // 5. x_proj: [BL,DI] x [96,DI]^T -> [BL,96]
  gemm_bf16_wmma<<<dim3(1, BL / 128), 256, 0, stream>>>(ucb, wx, ssm, BL, 96, DI);
  // 6. dt input extract
  extract_dt_kernel<<<blocks1d((long)BL * DTR), 256, 0, stream>>>(ssm, dti);
  // 7. dt_proj: [BL,DTR] x [DI,DTR]^T -> [BL,DI]
  gemm_bf16_wmma<<<dim3(DI / 256, BL / 128), 256, 0, stream>>>(dti, wdt, dtm, BL, DI, DTR);
  // 8. softplus(+bias)
  softplus_bias_kernel<<<blocks1d((long)BL * DI), 256, 0, stream>>>(dtm, dtp_b);
  // 9. selective scan + gate (y written over uc)
  mamba_scan_kernel<<<dim3(DI / 256, BB), 256, 0, stream>>>(dtm, uc, ssm, proj, A_log, Dv);
  // 10. y -> bf16
  f2bf_kernel<<<blocks1d((long)BL * DI), 256, 0, stream>>>(uc, yb, (long)BL * DI);
  // 11. out_proj: [BL,DI] x [DM,DI]^T -> [BL,DM]
  gemm_bf16_wmma<<<dim3(DM / 256, BL / 128), 256, 0, stream>>>(yb, wo, outm, BL, DM, DI);
  // 12. residual + final rmsnorm + head (last token only)
  final_head_kernel<<<BB, 256, 0, stream>>>(x, outm, normf_w, fc_w, fc_b, logits);
}
